// GraphWaveNet_62577673503858
// MI455X (gfx1250) — matrix operand order repaired and checked by reading, not compiled, not run on hardware
//
#include <hip/hip_runtime.h>
#include <math.h>

// ===================== types =====================
typedef __attribute__((ext_vector_type(16))) _Float16 v16h;
typedef __attribute__((ext_vector_type(8)))  float    v8f;
typedef __attribute__((ext_vector_type(4)))  unsigned int v4u;
typedef __attribute__((ext_vector_type(8)))  int v8i;
typedef __attribute__((ext_vector_type(4)))  int v4i;

// model constants
#define BB 16
#define NN 1024
#define RESC 32
#define SKIPC 256
#define ENDC 512
#define OUTD 12
#define NLAY 8
#define CATC 224

#if defined(__has_builtin)
#if __has_builtin(__builtin_amdgcn_tensor_load_to_lds) && __has_builtin(__builtin_amdgcn_s_wait_tensorcnt)
#define USE_TDM 1
#endif
#endif
#ifndef USE_TDM
#define USE_TDM 0
#endif

// LDS tile: 128 rows x 32 halves, row pitch 40 halves (64B payload + 16B TDM pad).
// 80B pitch => row bases 16B-aligned (ds_load_b128 fragments) and conflict-free banking:
// 20-dword stride, gcd(20,64)=4, 16 lanes cover all 64 banks exactly once.
#define TPITCH 40

#if USE_TDM
// Issue one TDM 2D tile load: 128 rows x 32 f16 elements, row stride = stride_elems, into LDS at lds_off.
// D# packing per CDNA5 ISA. Tiles are always fully in-bounds, so tensor dims
// are set large; pad: 4 DWORDs after every 16 DWORDs (pad_interval=3, pad_amount=3) => 80B LDS pitch.
// 6-arg builtin (clang-23 lane): (uint32x4 g0, int32x8 g1, int32x4 g2, int32x4 g3, int32x8 g4, i32 cpol)
__device__ __forceinline__ void tdm_issue(unsigned lds_off, const _Float16* gptr, unsigned stride_elems) {
    const unsigned long long ga = (unsigned long long)(size_t)gptr;
    v4u g0;
    g0[0] = 1u;                                            // count=1 (valid), user mode
    g0[1] = lds_off;                                       // lds_addr
    g0[2] = (unsigned)(ga & 0xFFFFFFFFu);                  // global_addr[31:0]
    g0[3] = (unsigned)((ga >> 32) & 0x1FFFFFFu) | (2u << 30); // global_addr[56:32] | type=2
    v8i g1;
    g1[0] = (int)((1u << 16) | (1u << 20) | (3u << 22) | (3u << 25)); // data_size=2B, pad_en, 16dw interval, 4dw pad
    g1[1] = 0;                                             // atomic_addr=0, tensor_dim0[15:0]=0
    g1[2] = (int)0x4000;                                   // tensor_dim0 = 1<<30 (no clip), tensor_dim1[15:0]=0
    g1[3] = (int)((32u << 16) | 0x10u);                    // tensor_dim1 = 1<<20, tile_dim0 = 32 elems
    g1[4] = 128;                                           // tile_dim1 = 128 rows, tile_dim2 = 0
    g1[5] = (int)stride_elems;                             // tensor_dim0_stride[31:0]
    g1[6] = 0;                                             // stride hi, dim1_stride lo
    g1[7] = 0;
    v4i g2 = {0, 0, 0, 0};
    v4i g3 = {0, 0, 0, 0};
    v8i g4 = {0, 0, 0, 0, 0, 0, 0, 0};
    __builtin_amdgcn_tensor_load_to_lds(g0, g1, g2, g3, g4, 0);
}
#endif

// ===================== WMMA GEMM (all-f16 operands, f32 accumulate) =====================
// C = A(MxK, f16, row-major lda) * Bt^T  where Bt is (NxK, f16, row-major ldb)
// epilogue: [+bias] [relu] ; optional f32 C (with += when beta) and/or f16 C.
// M%128==0, N%128==0, K%32==0. 256 threads = 8 waves (2x4 tile grid), double-buffered TDM staging.
__global__ __launch_bounds__(256)
void gemm_wmma_kernel(const _Float16* __restrict__ A, int lda,
                      const _Float16* __restrict__ Bt, int ldb,
                      float* __restrict__ C32, int ldc32,
                      _Float16* __restrict__ C16, int ldc16,
                      int M, int K, int N,
                      const float* __restrict__ bias,
                      int relu_out, int beta)
{
    __shared__ _Float16 AsB[2][128 * TPITCH];
    __shared__ _Float16 BsB[2][128 * TPITCH];

    const int tid  = threadIdx.x;
    const int lane = tid & 31;
    const int wave = tid >> 5;
    const int bm = blockIdx.x * 128;
    const int bn = blockIdx.y * 128;
    const int wm = (wave & 1) * 64;
    const int wn = (wave >> 1) * 32;

    // opaque never-taken store: keeps the optimizer from treating TDM-written LDS as undef
    if (M < 0) { AsB[0][tid] = (_Float16)0.f; BsB[0][tid] = (_Float16)0.f; }

    const v8f vzero = {0.f,0.f,0.f,0.f,0.f,0.f,0.f,0.f};
    v8f acc[4][2];
    #pragma unroll
    for (int i = 0; i < 4; ++i)
        #pragma unroll
        for (int j = 0; j < 2; ++j) acc[i][j] = vzero;

#if USE_TDM
    if (tid < 32) {   // wave 0 drives the Tensor Data Mover
        tdm_issue((unsigned)(size_t)&AsB[0][0], A  + (size_t)bm * lda, (unsigned)lda);
        tdm_issue((unsigned)(size_t)&BsB[0][0], Bt + (size_t)bn * ldb, (unsigned)ldb);
    }
#endif

    int cur = 0;
    for (int k0 = 0; k0 < K; k0 += 32) {
        const bool more = (k0 + 32) < K;
#if USE_TDM
        if (tid < 32) {
            if (more) {
                tdm_issue((unsigned)(size_t)&AsB[cur ^ 1][0], A  + (size_t)bm * lda + k0 + 32, (unsigned)lda);
                tdm_issue((unsigned)(size_t)&BsB[cur ^ 1][0], Bt + (size_t)bn * ldb + k0 + 32, (unsigned)ldb);
                __builtin_amdgcn_s_wait_tensorcnt(2);   // current pair complete, next pair in flight
            } else {
                __builtin_amdgcn_s_wait_tensorcnt(0);
            }
        }
        __asm__ __volatile__("" ::: "memory");
#else
        {   // fallback staging: plain vector copies
            const int r  = tid >> 1;
            const int cb = (tid & 1) * 16;
            const _Float16* ap = A  + (size_t)(bm + r) * lda + k0 + cb;
            const _Float16* bp = Bt + (size_t)(bn + r) * ldb + k0 + cb;
            #pragma unroll
            for (int q = 0; q < 16; ++q) AsB[cur][r * TPITCH + cb + q] = ap[q];
            #pragma unroll
            for (int q = 0; q < 16; ++q) BsB[cur][r * TPITCH + cb + q] = bp[q];
        }
#endif
        __syncthreads();

        // fragments per documented 16-bit layouts:
        // A 16x32: lanes 0-15 -> M=lane, K=0..7 & 16..23 ; lanes 16-31 -> K=8..15 & 24..31
        // B 32x16: lanes 0-15 -> N=lane, K=0..7 & 16..23 ; lanes 16-31 -> K=8..15 & 24..31
        const _Float16* As = &AsB[cur][0];
        const _Float16* Bs = &BsB[cur][0];
        v16h afr[4], bfr[2];
        const int mlo  = lane & 15;
        const int kb8  = (lane >> 4) * 8;
        const int kb16 = (lane >> 4) * 16;
        #pragma unroll
        for (int tm = 0; tm < 4; ++tm) {
            const int base = (wm + tm * 16 + mlo) * TPITCH;
            #pragma unroll
            for (int j = 0; j < 8; ++j) afr[tm][j]     = As[base + kb8 + j];
            #pragma unroll
            for (int j = 0; j < 8; ++j) afr[tm][8 + j] = As[base + 16 + kb8 + j];
        }
        #pragma unroll
        for (int tn = 0; tn < 2; ++tn) {
            const int base = (wn + tn * 16 + mlo) * TPITCH;
            #pragma unroll
            for (int j = 0; j < 8; ++j) bfr[tn][j]     = Bs[base + kb8 + j];
            #pragma unroll
            for (int j = 0; j < 8; ++j) bfr[tn][8 + j] = Bs[base + 16 + kb8 + j];
        }
        #pragma unroll
        for (int tm = 0; tm < 4; ++tm)
            #pragma unroll
            for (int tn = 0; tn < 2; ++tn)
                acc[tm][tn] = __builtin_amdgcn_wmma_f32_16x16x32_f16(
                    false, afr[tm], false, bfr[tn], (short)0, acc[tm][tn], false, false);
        __syncthreads();   // all waves done reading before next tile overwrites
        cur ^= 1;
    }

    // epilogue: C/D f32 16x16 layout: lane<16 -> M=r, lane>=16 -> M=r+8, N=lane&15
    const int rbase = (lane >> 4) * 8;
    const int cl    = lane & 15;
    #pragma unroll
    for (int tm = 0; tm < 4; ++tm) {
        #pragma unroll
        for (int tn = 0; tn < 2; ++tn) {
            const int col = bn + wn + tn * 16 + cl;
            #pragma unroll
            for (int r = 0; r < 8; ++r) {
                const int row = bm + wm + tm * 16 + rbase + r;
                float v = acc[tm][tn][r];
                if (bias) v += bias[col];
                if (relu_out) v = fmaxf(v, 0.f);
                if (C32) {
                    float* cp = C32 + (size_t)row * ldc32 + col;
                    if (beta) v += *cp;
                    *cp = v;
                }
                if (C16) C16[(size_t)row * ldc16 + col] = (_Float16)v;
            }
        }
    }
}

// ===================== small kernels =====================

// out(N x K, f16) = transpose(in(K x N, f32))   [one-time weight/support prep]
__global__ void transpose_cvt_kernel(const float* __restrict__ in, _Float16* __restrict__ out,
                                     int K, int N) {
    int i = blockIdx.x * 256 + threadIdx.x;
    const int n = i / K, k = i % K;
    out[i] = (_Float16)in[(size_t)k * N + n];
}

// straight f32 -> f16
__global__ void cvt_kernel(const float* __restrict__ in, _Float16* __restrict__ out) {
    int i = blockIdx.x * 256 + threadIdx.x;
    out[i] = (_Float16)in[i];
}

// hidden slice (B*N x 96 from ld 192) -> f16
__global__ void slice_cvt_kernel(const float* __restrict__ in, int ld, int off,
                                 _Float16* __restrict__ out) {
    int i = blockIdx.x * 256 + threadIdx.x;
    const int m = i / 96, k = i % 96;
    out[i] = (_Float16)in[(size_t)m * ld + off + k];
}

// relu + f32 -> f16 (skip before end1)
__global__ void relu_cvt_kernel(const float* __restrict__ in, _Float16* __restrict__ out) {
    int i = blockIdx.x * 256 + threadIdx.x;
    out[i] = (_Float16)fmaxf(in[i], 0.f);
}

// adpT[w][v] = softmax_w(relu(nv1 @ nv2))[v][w], written transposed in f16; one block per row v
__global__ void adp_kernel(const float* __restrict__ nv1, const float* __restrict__ nv2,
                           _Float16* __restrict__ adpT) {
    const int v = blockIdx.x, tid = threadIdx.x;
    float nr[10];
    #pragma unroll
    for (int k = 0; k < 10; ++k) nr[k] = nv1[v * 10 + k];
    float z[4], lmax = -1e30f;
    #pragma unroll
    for (int r = 0; r < 4; ++r) {
        const int w = tid + r * 256;
        float a = 0.f;
        #pragma unroll
        for (int k = 0; k < 10; ++k) a += nr[k] * nv2[k * NN + w];
        a = fmaxf(a, 0.f);
        z[r] = a; lmax = fmaxf(lmax, a);
    }
    __shared__ float red[256];
    red[tid] = lmax; __syncthreads();
    for (int s = 128; s > 0; s >>= 1) { if (tid < s) red[tid] = fmaxf(red[tid], red[tid + s]); __syncthreads(); }
    const float m = red[0]; __syncthreads();
    float lsum = 0.f;
    #pragma unroll
    for (int r = 0; r < 4; ++r) { z[r] = __expf(z[r] - m); lsum += z[r]; }
    red[tid] = lsum; __syncthreads();
    for (int s = 128; s > 0; s >>= 1) { if (tid < s) red[tid] += red[tid + s]; __syncthreads(); }
    const float inv = 1.f / red[0];
    #pragma unroll
    for (int r = 0; r < 4; ++r) adpT[(size_t)(tid + r * 256) * NN + v] = (_Float16)(z[r] * inv);
}

// X0[b][t][c][n], t in [0,13): t==0 is the left pad; start 1x1 conv from x_in(B,12,N,2)
__global__ void start_kernel(const float* __restrict__ x_in, float* __restrict__ X,
                             const float* __restrict__ sw, const float* __restrict__ sb) {
    int i = blockIdx.x * 256 + threadIdx.x;
    const int n = i & 1023, c = (i >> 10) & 31, bt = i >> 15;
    const int t = bt % 13, b = bt / 13;
    float v = sb[c];
    if (t > 0) {
        const float* xi = x_in + (((size_t)(b * 12 + (t - 1)) * NN) + n) * 2;
        v += sw[c * 2] * xi[0] + sw[c * 2 + 1] * xi[1];
    }
    X[i] = v;
}

// G[b][t][o][n] (f16) = tanh(dconv_filt) * sigmoid(dconv_gate)
__global__ void dconv_kernel(const float* __restrict__ X, _Float16* __restrict__ G,
                             const float* __restrict__ fw, const float* __restrict__ fb,
                             const float* __restrict__ gw, const float* __restrict__ gb,
                             int Tin, int Tout, int d) {
    int i = blockIdx.x * 256 + threadIdx.x;
    const int n = i & 1023, o = (i >> 10) & 31, bt = i >> 15;
    const int t = bt % Tout, b = bt / Tout;
    const float* x0 = X + ((size_t)(b * Tin + t) * RESC) * NN + n;
    const float* x1 = X + ((size_t)(b * Tin + t + d) * RESC) * NN + n;
    float f = fb[o], g = gb[o];
    #pragma unroll 8
    for (int c = 0; c < 32; ++c) {
        const float a0 = x0[(size_t)c * NN], a1 = x1[(size_t)c * NN];
        f += fw[(o * 32 + c) * 2] * a0 + fw[(o * 32 + c) * 2 + 1] * a1;
        g += gw[(o * 32 + c) * 2] * a0 + gw[(o * 32 + c) * 2 + 1] * a1;
    }
    G[i] = (_Float16)(tanhf(f) * (1.f / (1.f + __expf(-g))));
}

// skip[b][n][o] (+)= skip_w @ G[:, Tout-1, :, :] + skip_b    (only the surviving last slice)
__global__ void skip_kernel(const _Float16* __restrict__ G, float* __restrict__ skip,
                            const float* __restrict__ sw, const float* __restrict__ sb,
                            int Tout, int init) {
    const int b = blockIdx.x >> 10, n = blockIdx.x & 1023, o = threadIdx.x;
    __shared__ float xs[32];
    if (o < 32) xs[o] = (float)G[((size_t)(b * Tout + (Tout - 1)) * RESC + o) * NN + n];
    __syncthreads();
    float v = sb[o];
    #pragma unroll
    for (int c = 0; c < 32; ++c) v += sw[o * 32 + c] * xs[c];
    const size_t idx = ((size_t)(b * NN + n)) * SKIPC + o;
    if (init) skip[idx] = v; else skip[idx] += v;
}

// newX = gcn_bias + residual trailing slice
__global__ void gcn_init_kernel(float* __restrict__ Y, const float* __restrict__ Xin,
                                const float* __restrict__ gb, int Tin, int Tout, int d) {
    int i = blockIdx.x * 256 + threadIdx.x;
    const int n = i & 1023, o = (i >> 10) & 31, bt = i >> 15;
    const int t = bt % Tout, b = bt / Tout;
    Y[i] = gb[o] + Xin[((size_t)(b * Tin + t + d) * RESC + o) * NN + n];
}

// newX[b][t][o][n] += sum_c W[o][base+c] * Z[b][t][c][n]   (gcn 1x1 channel mix, f16 activations)
__global__ void gcn_acc_kernel(float* __restrict__ Y, const _Float16* __restrict__ Z,
                               const float* __restrict__ gw, int base, int Tout) {
    int i = blockIdx.x * 256 + threadIdx.x;
    const int n = i & 1023, o = (i >> 10) & 31, bt = i >> 15;
    const _Float16* z = Z + ((size_t)bt * RESC) * NN + n;
    const float* w = gw + o * CATC + base;
    float acc = Y[i];
    #pragma unroll 8
    for (int c = 0; c < 32; ++c) acc += w[c] * (float)z[(size_t)c * NN];
    Y[i] = acc;
}

__global__ void zero_stats_kernel(float* s) { s[threadIdx.x] = 0.f; }

__global__ void bn_reduce_kernel(const float* __restrict__ X, int total, float* __restrict__ st) {
    __shared__ float s1[32], s2[32];
    const int tid = threadIdx.x;
    if (tid < 32) { s1[tid] = 0.f; s2[tid] = 0.f; }
    __syncthreads();
    for (int i = blockIdx.x * 256 + tid; i < total; i += gridDim.x * 256) {
        const int c = (i >> 10) & 31;
        const float v = X[i];
        atomicAdd(&s1[c], v);
        atomicAdd(&s2[c], v * v);
    }
    __syncthreads();
    if (tid < 32) { atomicAdd(&st[tid], s1[tid]); atomicAdd(&st[32 + tid], s2[tid]); }
}

__global__ void bn_apply_kernel(float* __restrict__ X, const float* __restrict__ st,
                                const float* __restrict__ g, const float* __restrict__ b,
                                float inv_cnt) {
    int i = blockIdx.x * 256 + threadIdx.x;
    const int c = (i >> 10) & 31;
    const float mu  = st[c] * inv_cnt;
    const float var = st[32 + c] * inv_cnt - mu * mu;
    X[i] = (X[i] - mu) * rsqrtf(var + 1e-5f) * g[c] + b[c];
}

// out[b][n][o] = end2_b[o] + end2_w[o] . H1[m]   (K=512, N=12)
__global__ void end2_kernel(const _Float16* __restrict__ H1, const float* __restrict__ W,
                            const float* __restrict__ bias, float* __restrict__ out) {
    int i = blockIdx.x * 256 + threadIdx.x;
    if (i >= BB * NN * OUTD) return;
    const int m = i / OUTD, o = i % OUTD;
    const _Float16* h = H1 + (size_t)m * ENDC;
    const float* w = W + o * ENDC;
    float acc = bias[o];
    #pragma unroll 8
    for (int k = 0; k < ENDC; ++k) acc += w[k] * (float)h[k];
    out[i] = acc;
}

// ===================== host orchestration =====================
extern "C" void kernel_launch(void* const* d_in, const int* in_sizes, int n_in,
                              void* d_out, int out_size, void* d_ws, size_t ws_size,
                              hipStream_t stream) {
    (void)in_sizes; (void)n_in; (void)out_size; (void)ws_size;
    const float* x_in    = (const float*)d_in[0];
    const float* hidden  = (const float*)d_in[1];
    const float* sup     = (const float*)d_in[2];
    const float* start_w = (const float*)d_in[3];
    const float* start_b = (const float*)d_in[4];
    const float* nv1     = (const float*)d_in[5];
    const float* nv2     = (const float*)d_in[6];
    const float* filt_w  = (const float*)d_in[7];
    const float* filt_b  = (const float*)d_in[8];
    const float* gate_w  = (const float*)d_in[9];
    const float* gate_b  = (const float*)d_in[10];
    const float* skip_w  = (const float*)d_in[11];
    const float* skip_b  = (const float*)d_in[12];
    const float* gcn_w   = (const float*)d_in[13];
    const float* gcn_b   = (const float*)d_in[14];
    const float* bn_g    = (const float*)d_in[15];
    const float* bn_b    = (const float*)d_in[16];
    const float* t_w1 = (const float*)d_in[17]; const float* t_b1 = (const float*)d_in[18];
    const float* t_w2 = (const float*)d_in[19]; const float* t_b2 = (const float*)d_in[20];
    const float* s_w1 = (const float*)d_in[21]; const float* s_b1 = (const float*)d_in[22];
    const float* s_w2 = (const float*)d_in[23]; const float* s_b2 = (const float*)d_in[24];
    const float* end1_w = (const float*)d_in[25]; const float* end1_b = (const float*)d_in[26];
    const float* end2_w = (const float*)d_in[27]; const float* end2_b = (const float*)d_in[28];

    char* ws = (char*)d_ws;
    size_t off = 0;
    auto take = [&](size_t bytes) { size_t o = off; off += (bytes + 255) & ~(size_t)255; return o; };

    _Float16* supT = (_Float16*)(ws + take((size_t)3 * NN * NN * 2));              // transposed f16 supports+adp
    float* Xa   = (float*)(ws + take((size_t)BB * 13 * RESC * NN * 4));
    float* Xb   = (float*)(ws + take((size_t)BB * 13 * RESC * NN * 4));
    _Float16* Gh  = (_Float16*)(ws + take((size_t)BB * 12 * RESC * NN * 2));
    _Float16* Z1h = (_Float16*)(ws + take((size_t)BB * 12 * RESC * NN * 2));
    _Float16* Z2h = (_Float16*)(ws + take((size_t)BB * 12 * RESC * NN * 2));
    float* skip = (float*)(ws + take((size_t)BB * NN * SKIPC * 4));
    _Float16* Hh0 = (_Float16*)(ws + take((size_t)BB * NN * 96 * 2));
    _Float16* Hh1 = (_Float16*)(ws + take((size_t)BB * NN * 96 * 2));
    _Float16* H1h = (_Float16*)(ws + take((size_t)BB * NN * ENDC * 2));
    _Float16* Skh = (_Float16*)(ws + take((size_t)BB * NN * SKIPC * 2));
    _Float16* t1T = (_Float16*)(ws + take((size_t)512 * 96 * 2));
    _Float16* t2T = (_Float16*)(ws + take((size_t)256 * 512 * 2));
    _Float16* s1T = (_Float16*)(ws + take((size_t)512 * 96 * 2));
    _Float16* s2T = (_Float16*)(ws + take((size_t)256 * 512 * 2));
    _Float16* e1h = (_Float16*)(ws + take((size_t)512 * 256 * 2));
    float* st   = (float*)(ws + take(256));

    // ---- one-time prep: transposed f16 supports, adaptive adjacency, weights, start conv ----
    transpose_cvt_kernel<<<(NN * NN) / 256, 256, 0, stream>>>(sup, supT, NN, NN);
    transpose_cvt_kernel<<<(NN * NN) / 256, 256, 0, stream>>>(sup + (size_t)NN * NN, supT + (size_t)NN * NN, NN, NN);
    adp_kernel<<<NN, 256, 0, stream>>>(nv1, nv2, supT + (size_t)2 * NN * NN);
    transpose_cvt_kernel<<<(512 * 96) / 256, 256, 0, stream>>>(t_w1, t1T, 96, 512);
    transpose_cvt_kernel<<<(256 * 512) / 256, 256, 0, stream>>>(t_w2, t2T, 512, 256);
    transpose_cvt_kernel<<<(512 * 96) / 256, 256, 0, stream>>>(s_w1, s1T, 96, 512);
    transpose_cvt_kernel<<<(256 * 512) / 256, 256, 0, stream>>>(s_w2, s2T, 512, 256);
    cvt_kernel<<<(512 * 256) / 256, 256, 0, stream>>>(end1_w, e1h);   // (o,c) already Bt orientation
    slice_cvt_kernel<<<(BB * NN * 96) / 256, 256, 0, stream>>>(hidden, 192, 0, Hh0);
    slice_cvt_kernel<<<(BB * NN * 96) / 256, 256, 0, stream>>>(hidden, 192, 96, Hh1);
    start_kernel<<<(BB * 13 * RESC * NN) / 256, 256, 0, stream>>>(x_in, Xa, start_w, start_b);

    // ---- layers ----
    const int dils[NLAY] = {1, 2, 1, 2, 1, 2, 1, 2};
    float* Xin = Xa; float* Xout = Xb;
    int Tin = 13;
    for (int i = 0; i < NLAY; ++i) {
        const int d = dils[i], Tout = Tin - d;
        const int nG = BB * Tout * RESC * NN;
        const int M  = BB * Tout * RESC;      // multiple of 128

        dconv_kernel<<<nG / 256, 256, 0, stream>>>(Xin, Gh,
            filt_w + (size_t)i * 2048, filt_b + i * 32,
            gate_w + (size_t)i * 2048, gate_b + i * 32, Tin, Tout, d);
        skip_kernel<<<BB * NN, SKIPC, 0, stream>>>(Gh, skip,
            skip_w + (size_t)i * SKIPC * 32, skip_b + i * SKIPC, Tout, i == 0);

        if (i < NLAY - 1) {   // last layer's x output is never consumed
            gcn_init_kernel<<<nG / 256, 256, 0, stream>>>(Xout, Xin, gcn_b + i * 32, Tin, Tout, d);
            gcn_acc_kernel<<<nG / 256, 256, 0, stream>>>(Xout, Gh, gcn_w + (size_t)i * RESC * CATC, 0, Tout);
            for (int s = 0; s < 3; ++s) {
                const _Float16* At = supT + (size_t)s * NN * NN;
                gemm_wmma_kernel<<<dim3(M / 128, NN / 128), 256, 0, stream>>>(
                    Gh, NN, At, NN, nullptr, 0, Z1h, NN, M, NN, NN, nullptr, 0, 0);
                gcn_acc_kernel<<<nG / 256, 256, 0, stream>>>(Xout, Z1h,
                    gcn_w + (size_t)i * RESC * CATC, (1 + 2 * s) * 32, Tout);
                gemm_wmma_kernel<<<dim3(M / 128, NN / 128), 256, 0, stream>>>(
                    Z1h, NN, At, NN, nullptr, 0, Z2h, NN, M, NN, NN, nullptr, 0, 0);
                gcn_acc_kernel<<<nG / 256, 256, 0, stream>>>(Xout, Z2h,
                    gcn_w + (size_t)i * RESC * CATC, (2 + 2 * s) * 32, Tout);
            }
            zero_stats_kernel<<<1, 64, 0, stream>>>(st);
            bn_reduce_kernel<<<1024, 256, 0, stream>>>(Xout, nG, st);
            bn_apply_kernel<<<nG / 256, 256, 0, stream>>>(Xout, st, bn_g + i * 32, bn_b + i * 32,
                                                          1.f / ((float)BB * NN * Tout));
            float* tmp = Xin; Xin = Xout; Xout = tmp;
            Tin = Tout;
        }
    }

    // ---- hidden-state MLPs, accumulated into skip (layout [b][n][256] == ldc 256) ----
    const int Mh = BB * NN;   // 16384
    gemm_wmma_kernel<<<dim3(Mh / 128, 512 / 128), 256, 0, stream>>>(
        Hh0, 96, t1T, 96, nullptr, 0, H1h, 512, Mh, 96, 512, t_b1, 1, 0);
    gemm_wmma_kernel<<<dim3(Mh / 128, SKIPC / 128), 256, 0, stream>>>(
        H1h, 512, t2T, 512, skip, SKIPC, nullptr, 0, Mh, 512, SKIPC, t_b2, 1, 1);
    gemm_wmma_kernel<<<dim3(Mh / 128, 512 / 128), 256, 0, stream>>>(
        Hh1, 96, s1T, 96, nullptr, 0, H1h, 512, Mh, 96, 512, s_b1, 1, 0);
    gemm_wmma_kernel<<<dim3(Mh / 128, SKIPC / 128), 256, 0, stream>>>(
        H1h, 512, s2T, 512, skip, SKIPC, nullptr, 0, Mh, 512, SKIPC, s_b2, 1, 1);

    // ---- end convs ----
    relu_cvt_kernel<<<(Mh * SKIPC) / 256, 256, 0, stream>>>(skip, Skh);
    gemm_wmma_kernel<<<dim3(Mh / 128, ENDC / 128), 256, 0, stream>>>(
        Skh, SKIPC, e1h, SKIPC, nullptr, 0, H1h, ENDC, Mh, SKIPC, ENDC, end1_b, 1, 0);
    end2_kernel<<<(BB * NN * OUTD + 255) / 256, 256, 0, stream>>>(H1h, end2_w, end2_b, (float*)d_out);
}
